// QuantLlamaAttention_833223655809
// MI455X (gfx1250) — 8-bit matrix operand order repaired and checked by reading, compiled, not run on hardware
//
#include <hip/hip_runtime.h>
#include <cstdint>

#define BB   2
#define SS   1024
#define HH   4096
#define NHH  32
#define DD   128
#define MTOK (BB * SS)

typedef __attribute__((ext_vector_type(8)))  int      v8i;
typedef __attribute__((ext_vector_type(8)))  float    v8f;
typedef __attribute__((ext_vector_type(16))) _Float16 v16h;

union FragI8 { unsigned long long u64[4]; v8i v; };

#if defined(__has_builtin)
#if __has_builtin(__builtin_amdgcn_tensor_load_to_lds) && __has_builtin(__builtin_amdgcn_s_wait_tensorcnt)
#define HAVE_TDM 1
#endif
#endif
#ifndef HAVE_TDM
#define HAVE_TDM 0
#endif

#if HAVE_TDM
typedef unsigned int tdm_u4 __attribute__((ext_vector_type(4)));
typedef int          tdm_i8 __attribute__((ext_vector_type(8)));
typedef int          tdm_i4 __attribute__((ext_vector_type(4)));

__device__ __forceinline__ void tdm_load_tile_2d(const signed char* gsrc,
                                                 unsigned ldsOff,
                                                 unsigned tensorK, unsigned tensorRows,
                                                 unsigned strideK,
                                                 unsigned tileK, unsigned tileRows) {
  unsigned long long ga = (unsigned long long)(uintptr_t)gsrc;
  tdm_u4 g0;
  g0[0] = 1u;
  g0[1] = ldsOff;
  g0[2] = (unsigned)(ga & 0xFFFFFFFFu);
  g0[3] = (unsigned)((ga >> 32) & 0x01FFFFFFu)
        | (2u << 30);
  tdm_i8 g1;
  g1[0] = 0;
  g1[1] = (int)((tensorK & 0xFFFFu) << 16);
  g1[2] = (int)((tensorK >> 16) | ((tensorRows & 0xFFFFu) << 16));
  g1[3] = (int)((tensorRows >> 16) | (tileK << 16));
  g1[4] = (int)(tileRows & 0xFFFFu);
  g1[5] = (int)strideK;
  g1[6] = 0;
  g1[7] = 0;
  tdm_i4 z4 = {};
#if __clang_major__ >= 23
  tdm_i8 z8 = {};
  __builtin_amdgcn_tensor_load_to_lds(g0, g1, z4, z4, z8, 0);
#else
  __builtin_amdgcn_tensor_load_to_lds(g0, g1, z4, z4, 0);
#endif
}
#endif

__global__ void rowquant_kernel(const float* __restrict__ src,
                                signed char* __restrict__ dst,
                                float* __restrict__ scales,
                                int cols, float qmax) {
  const int row = blockIdx.x;
  const float* p = src + (size_t)row * cols;
  float m = 0.f;
  for (int c = threadIdx.x; c < cols; c += blockDim.x) m = fmaxf(m, fabsf(p[c]));
  __shared__ float red[256];
  red[threadIdx.x] = m; __syncthreads();
  for (int s = 128; s > 0; s >>= 1) {
    if ((int)threadIdx.x < s) red[threadIdx.x] = fmaxf(red[threadIdx.x], red[threadIdx.x + s]);
    __syncthreads();
  }
  const float scale = fmaxf(red[0] / qmax, 1e-8f);
  const float inv = 1.f / scale;
  if (threadIdx.x == 0) scales[row] = scale;
  signed char* q = dst + (size_t)row * cols;
  const float lo = -qmax - 1.f, hi = qmax;
  for (int c = threadIdx.x; c < cols; c += blockDim.x) {
    float v = rintf(p[c] * inv);
    q[c] = (signed char)fminf(fmaxf(v, lo), hi);
  }
}

__global__ void gemm_i8_kernel(const signed char* __restrict__ A,
                               const float* __restrict__ sA,
                               const signed char* __restrict__ Bm,
                               const float* __restrict__ sB,
                               float* __restrict__ C,
                               int M, int N, int K) {
  __shared__ signed char smA[2][128 * 64];
  __shared__ signed char smB[2][128 * 64];

  const int w    = threadIdx.x >> 5;
  const int lane = threadIdx.x & 31;
  const int waveM = w >> 1;
  const int waveN = w & 1;
  const int bm0 = blockIdx.y * 128;
  const int bn0 = blockIdx.x * 128;
  const int mrow = lane & 15;
  const int kh   = lane >> 4;

  const int KT = K / 64;
  v8i acc[2][4] = {};

#if HAVE_TDM
  if (w == 0) {
    tdm_load_tile_2d(A  + (size_t)bm0 * K, (unsigned)(uintptr_t)&smA[0][0],
                     (unsigned)K, (unsigned)M, (unsigned)K, 64u, 128u);
    tdm_load_tile_2d(Bm + (size_t)bn0 * K, (unsigned)(uintptr_t)&smB[0][0],
                     (unsigned)K, (unsigned)N, (unsigned)K, 64u, 128u);
  }
#endif

  for (int kt = 0; kt < KT; ++kt) {
    const int buf = kt & 1;
#if HAVE_TDM
    const bool more = (kt + 1) < KT;
    if (w == 0) {
      if (more) {
        const int k1 = (kt + 1) * 64;
        tdm_load_tile_2d(A  + (size_t)bm0 * K + k1, (unsigned)(uintptr_t)&smA[buf ^ 1][0],
                         (unsigned)K, (unsigned)M, (unsigned)K, 64u, 128u);
        tdm_load_tile_2d(Bm + (size_t)bn0 * K + k1, (unsigned)(uintptr_t)&smB[buf ^ 1][0],
                         (unsigned)K, (unsigned)N, (unsigned)K, 64u, 128u);
        __builtin_amdgcn_s_wait_tensorcnt(2);
      } else {
        __builtin_amdgcn_s_wait_tensorcnt(0);
      }
    }
    __syncthreads();
#else
    __syncthreads();
    {
      const int k0 = kt * 64;
      const int t = threadIdx.x;
      const int r = t >> 1, half = (t & 1) * 32;
      *(uint4*)&smA[buf][r * 64 + half] =
          *(const uint4*)(A + (size_t)(bm0 + r) * K + k0 + half);
      *(uint4*)&smA[buf][r * 64 + half + 16] =
          *(const uint4*)(A + (size_t)(bm0 + r) * K + k0 + half + 16);
      *(uint4*)&smB[buf][r * 64 + half] =
          *(const uint4*)(Bm + (size_t)(bn0 + r) * K + k0 + half);
      *(uint4*)&smB[buf][r * 64 + half + 16] =
          *(const uint4*)(Bm + (size_t)(bn0 + r) * K + k0 + half + 16);
    }
    __syncthreads();
#endif

    FragI8 a[2], b[4];
#pragma unroll
    for (int ti = 0; ti < 2; ++ti) {
      const signed char* ap = &smA[buf][(waveM * 32 + ti * 16 + mrow) * 64 + kh * 16];
#pragma unroll
      for (int c = 0; c < 4; ++c)
        a[ti].u64[c] = *(const unsigned long long*)(ap + (c & 1) * 8 + (c >> 1) * 32);
    }
#pragma unroll
    for (int tj = 0; tj < 4; ++tj) {
      const signed char* bp = &smB[buf][(waveN * 64 + tj * 16 + mrow) * 64 + kh * 16];
      b[tj].u64[0] = *(const unsigned long long*)(bp);
      b[tj].u64[1] = *(const unsigned long long*)(bp + 8);
      b[tj].u64[2] = *(const unsigned long long*)(bp + 32);
      b[tj].u64[3] = *(const unsigned long long*)(bp + 40);
    }
#pragma unroll
    for (int ti = 0; ti < 2; ++ti)
#pragma unroll
      for (int tj = 0; tj < 4; ++tj)
        acc[ti][tj] = __builtin_amdgcn_wmma_i32_16x16x64_iu8(
            true, a[ti].v, true, b[tj].v, acc[ti][tj], false, false);

    __syncthreads();
  }

#pragma unroll
  for (int ti = 0; ti < 2; ++ti) {
#pragma unroll
    for (int tj = 0; tj < 4; ++tj) {
      const int col = bn0 + waveN * 64 + tj * 16 + mrow;
      const float sb = sB[col];
#pragma unroll
      for (int r = 0; r < 8; ++r) {
        const int row = bm0 + waveM * 32 + ti * 16 + r + 8 * kh;
        C[(size_t)row * N + col] = sA[row] * sb * (float)acc[ti][tj][r];
      }
    }
  }
}

__global__ void rope_quant_kernel(const float* __restrict__ qbuf,
                                  const float* __restrict__ kbuf,
                                  const float* __restrict__ vbuf,
                                  signed char* __restrict__ q8, float* __restrict__ sq,
                                  signed char* __restrict__ k8, float* __restrict__ sk,
                                  signed char* __restrict__ v8, float* __restrict__ sv) {
  const int r  = blockIdx.x;
  const int bh = r / SS, s = r % SS;
  const int b  = bh / NHH, h = bh % NHH;
  const int d  = threadIdx.x;
  const size_t base = ((size_t)(b * SS + s)) * HH + h * DD;
  const int dm = d & 63;
  const float inv_freq = __expf(-(float)(2 * dm) * (1.0f / (float)DD) * 9.210340371976184f);
  const float ang = (float)s * inv_freq;
  float sn, cs; __sincosf(ang, &sn, &cs);

  const float xq = qbuf[base + d], xk = kbuf[base + d];
  const int   od = (d < 64) ? d + 64 : d - 64;
  const float oq = qbuf[base + od], ok = kbuf[base + od];
  const float rq = (d < 64) ? (xq * cs - oq * sn) : (xq * cs + oq * sn);
  const float rk = (d < 64) ? (xk * cs - ok * sn) : (xk * cs + ok * sn);
  const float xv = vbuf[base + d];

  __shared__ float red[128];
  auto quant_row = [&](float val, signed char* dst, float* sc) {
    red[d] = fabsf(val); __syncthreads();
    for (int st = 64; st > 0; st >>= 1) {
      if (d < st) red[d] = fmaxf(red[d], red[d + st]);
      __syncthreads();
    }
    const float scale = fmaxf(red[0] / 127.f, 1e-8f);
    __syncthreads();
    float qv = fminf(fmaxf(rintf(val / scale), -128.f), 127.f);
    dst[(size_t)r * DD + d] = (signed char)qv;
    if (d == 0) sc[r] = scale;
  };
  quant_row(rq, q8, sq);
  quant_row(rk, k8, sk);
  quant_row(xv, v8, sv);
}

__global__ void attention_kernel(const signed char* __restrict__ q8, const float* __restrict__ sq,
                                 const signed char* __restrict__ k8, const float* __restrict__ sk,
                                 const signed char* __restrict__ v8, const float* __restrict__ sv,
                                 float* __restrict__ attn) {
  extern __shared__ char dynsm[];
  float* sc = (float*)dynsm;
  float* sp = sc + 16 * SS;

  const int qt = blockIdx.x, bh = blockIdx.y;
  const int b  = bh / NHH, h = bh % NHH;
  const int q0 = qt * 16;
  const int w    = threadIdx.x >> 5;
  const int lane = threadIdx.x & 31;
  const int mrow = lane & 15, kh = lane >> 4;

  const int ntiles  = qt + 1;
  const int numk    = ntiles * 16;
  const int numkpad = (numk + 31) & ~31;

  const signed char* qp_ = q8 + ((size_t)bh * SS + (q0 + mrow)) * DD;
  FragI8 aq0, aq1;
#pragma unroll
  for (int c = 0; c < 4; ++c) {
    aq0.u64[c] = *(const unsigned long long*)(qp_ + kh * 16 + (c & 1) * 8 + (c >> 1) * 32);
    aq1.u64[c] = *(const unsigned long long*)(qp_ + 64 + kh * 16 + (c & 1) * 8 + (c >> 1) * 32);
  }

  const float rsqrtD = 0.08838834764831843f;
  for (int j = w; j < ntiles; j += 8) {
    const int kglob = j * 16 + mrow;
    const signed char* kp = k8 + ((size_t)bh * SS + kglob) * DD;
    FragI8 b0, b1;
    b0.u64[0] = *(const unsigned long long*)(kp + kh * 16);
    b0.u64[1] = *(const unsigned long long*)(kp + kh * 16 + 8);
    b0.u64[2] = *(const unsigned long long*)(kp + 32 + kh * 16);
    b0.u64[3] = *(const unsigned long long*)(kp + 32 + kh * 16 + 8);
    b1.u64[0] = *(const unsigned long long*)(kp + 64 + kh * 16);
    b1.u64[1] = *(const unsigned long long*)(kp + 64 + kh * 16 + 8);
    b1.u64[2] = *(const unsigned long long*)(kp + 96 + kh * 16);
    b1.u64[3] = *(const unsigned long long*)(kp + 96 + kh * 16 + 8);
    v8i z = {};
    v8i acc = __builtin_amdgcn_wmma_i32_16x16x64_iu8(true, aq0.v, true, b0.v, z,   false, false);
    acc     = __builtin_amdgcn_wmma_i32_16x16x64_iu8(true, aq1.v, true, b1.v, acc, false, false);
    const float skv = sk[(size_t)bh * SS + kglob];
#pragma unroll
    for (int r = 0; r < 8; ++r) {
      const int qrow = r + 8 * kh;
      const int qg   = q0 + qrow;
      float s = (float)acc[r] * sq[(size_t)bh * SS + qg] * skv * rsqrtD;
      if (kglob > qg) s = -1e30f;
      sc[qrow * SS + kglob] = s;
    }
  }
  __syncthreads();

  for (int rrow = w; rrow < 16; rrow += 8) {
    float m = -1e30f;
    for (int k = lane; k < numk; k += 32) m = fmaxf(m, sc[rrow * SS + k]);
#pragma unroll
    for (int off = 16; off > 0; off >>= 1) m = fmaxf(m, __shfl_xor(m, off, 32));
    float zsum = 0.f;
    for (int k = lane; k < numkpad; k += 32) {
      if (k < numk) {
        float e = __expf(sc[rrow * SS + k] - m);
        zsum += e;
        sc[rrow * SS + k] = rintf(127.f * e);
      } else {
        sc[rrow * SS + k] = 0.f;
      }
    }
#pragma unroll
    for (int off = 16; off > 0; off >>= 1) zsum += __shfl_xor(zsum, off, 32);
    if (lane == 0) sp[rrow] = 1.f / (127.f * zsum);
  }
  __syncthreads();

  const int dt = w;
  v8f facc = {};
  for (int kc = 0; kc < numkpad; kc += 32) {
    v16h a, bf;
#pragma unroll
    for (int e = 0; e < 16; ++e) {
      const int K  = (e < 8) ? (kh * 8 + e) : (16 + kh * 8 + (e - 8));
      const int kk = kc + K;
      a[e] = (_Float16)sc[mrow * SS + kk];
      const int dcol = dt * 16 + mrow;
      float vv = 0.f;
      if (kk < numk)
        vv = sv[(size_t)bh * SS + kk] * (float)v8[((size_t)bh * SS + kk) * DD + dcol];
      bf[e] = (_Float16)vv;
    }
    facc = __builtin_amdgcn_wmma_f32_16x16x32_f16(false, a, false, bf,
                                                  (short)0, facc, false, false);
  }
  const int d = dt * 16 + mrow;
#pragma unroll
  for (int r = 0; r < 8; ++r) {
    const int qrow = r + 8 * kh;
    const int qg   = q0 + qrow;
    attn[((size_t)(b * SS + qg)) * HH + h * DD + d] = facc[r] * sp[qrow];
  }
}

extern "C" void kernel_launch(void* const* d_in, const int* in_sizes, int n_in,
                              void* d_out, int out_size, void* d_ws, size_t ws_size,
                              hipStream_t stream) {
  (void)in_sizes; (void)n_in; (void)out_size; (void)ws_size;
  const float* hid = (const float*)d_in[0];
  const float* w_q = (const float*)d_in[1];
  const float* w_k = (const float*)d_in[2];
  const float* w_v = (const float*)d_in[3];
  const float* w_o = (const float*)d_in[4];
  float* out = (float*)d_out;

  char* ws = (char*)d_ws;
  size_t off = 0;
  auto alloc = [&](size_t bytes) -> char* {
    char* p = ws + off;
    off = (off + bytes + 255) & ~(size_t)255;
    return p;
  };
  signed char* x8   = (signed char*)alloc((size_t)MTOK * HH);
  float*       sx   = (float*)      alloc((size_t)MTOK * 4);
  signed char* w8[4]; float* sw[4];
  for (int i = 0; i < 4; ++i) {
    w8[i] = (signed char*)alloc((size_t)HH * HH);
    sw[i] = (float*)      alloc((size_t)HH * 4);
  }
  float* qbuf = (float*)alloc((size_t)MTOK * HH * 4);
  float* kbuf = (float*)alloc((size_t)MTOK * HH * 4);
  float* vbuf = (float*)alloc((size_t)MTOK * HH * 4);
  const size_t R = (size_t)BB * NHH * SS;
  signed char* q8a = (signed char*)alloc(R * DD);
  signed char* k8a = (signed char*)alloc(R * DD);
  signed char* v8a = (signed char*)alloc(R * DD);
  float* sqa = (float*)alloc(R * 4);
  float* ska = (float*)alloc(R * 4);
  float* sva = (float*)alloc(R * 4);

  const float* wsrc[4] = { w_q, w_k, w_v, w_o };

  rowquant_kernel<<<MTOK, 256, 0, stream>>>(hid, x8, sx, HH, 127.f);
  for (int i = 0; i < 4; ++i)
    rowquant_kernel<<<HH, 256, 0, stream>>>(wsrc[i], w8[i], sw[i], HH, 7.f);

  dim3 ggrid(HH / 128, MTOK / 128);
  gemm_i8_kernel<<<ggrid, 256, 0, stream>>>(x8, sx, w8[0], sw[0], qbuf, MTOK, HH, HH);
  gemm_i8_kernel<<<ggrid, 256, 0, stream>>>(x8, sx, w8[1], sw[1], kbuf, MTOK, HH, HH);
  gemm_i8_kernel<<<ggrid, 256, 0, stream>>>(x8, sx, w8[2], sw[2], vbuf, MTOK, HH, HH);

  rope_quant_kernel<<<(unsigned)R, 128, 0, stream>>>(qbuf, kbuf, vbuf,
                                                     q8a, sqa, k8a, ska, v8a, sva);

  float* attnbuf = qbuf;
  dim3 agrid(SS / 16, BB * NHH);
  const size_t dynls = (size_t)16 * SS * 4 + 16 * 4;
  attention_kernel<<<agrid, 256, dynls, stream>>>(q8a, sqa, k8a, ska, v8a, sva, attnbuf);

  rowquant_kernel<<<MTOK, 256, 0, stream>>>(attnbuf, x8, sx, HH, 127.f);
  gemm_i8_kernel<<<ggrid, 256, 0, stream>>>(x8, sx, w8[3], sw[3], out, MTOK, HH, HH);
}
